// MultiHeadAttention_67207648248315
// MI455X (gfx1250) — matrix operand order repaired and checked by reading, not compiled, not run on hardware
//
#include <hip/hip_runtime.h>
#include <hip/hip_bf16.h>

typedef __attribute__((ext_vector_type(16))) _Float16 v16h;
typedef __attribute__((ext_vector_type(8)))  _Float16 v8h;
typedef __attribute__((ext_vector_type(8)))  float    v8f;
typedef __attribute__((ext_vector_type(4)))  float    v4f;

#define DM   1024   // d_model
#define NH   16     // heads
#define DK   64     // head dim
#define BB   4      // batch
#define SS   2048   // seq
#define MROWS (BB*SS)   // 8192 token rows
#define MF   2      // 16-row fragments per wave in the fused attention kernel

#define WMMA_F16(a, b, c) \
  __builtin_amdgcn_wmma_f32_16x16x32_f16(false, (a), false, (b), (short)0, (c), false, false)

static __device__ __forceinline__ int lane_id() { return (int)(threadIdx.x & 31u); }

// ---- Fragment loaders (ISA 7.12.2 layouts, wave32) ----------------------

// A-matrix 16x32 f16 fragment from an f32 row-major source (convert in-flight).
static __device__ __forceinline__ v16h load_a_f32(const float* A, int lda, int m0, int k0) {
  const int lane = lane_id();
  const int m  = m0 + (lane & 15);
  const int kb = k0 + ((lane >> 4) << 3);
  const float* p = A + (size_t)m * lda + kb;
  v4f a0 = *(const v4f*)(p);
  v4f a1 = *(const v4f*)(p + 4);
  v4f a2 = *(const v4f*)(p + 16);
  v4f a3 = *(const v4f*)(p + 20);
  v16h r;
#pragma unroll
  for (int i = 0; i < 4; ++i) {
    r[i]      = (_Float16)a0[i];
    r[4 + i]  = (_Float16)a1[i];
    r[8 + i]  = (_Float16)a2[i];
    r[12 + i] = (_Float16)a3[i];
  }
  return r;
}

// A-matrix 16x32 f16 fragment from an f16 row-major source (two b128 loads).
static __device__ __forceinline__ v16h load_a_f16(const _Float16* A, int lda, int m0, int k0) {
  const int lane = lane_id();
  const int m  = m0 + (lane & 15);
  const int kb = k0 + ((lane >> 4) << 3);
  const _Float16* p = A + (size_t)m * lda + kb;
  v8h a0 = *(const v8h*)(p);
  v8h a1 = *(const v8h*)(p + 16);
  v16h r;
#pragma unroll
  for (int i = 0; i < 8; ++i) { r[i] = a0[i]; r[8 + i] = a1[i]; }
  return r;
}

// A-matrix 16x32 f16 fragment from a per-wave LDS tile with row stride 40 halves.
static __device__ __forceinline__ v16h load_a_lds(const _Float16* tb) {
  const int lane = lane_id();
  const _Float16* p = tb + (lane & 15) * 40 + ((lane >> 4) << 3);
  v8h a0 = *(const v8h*)(p);
  v8h a1 = *(const v8h*)(p + 16);
  v16h r;
#pragma unroll
  for (int i = 0; i < 8; ++i) { r[i] = a0[i]; r[8 + i] = a1[i]; }
  return r;
}

// B-matrix 32x16 f16 fragment. Source row n (K-contiguous) becomes output column n.
static __device__ __forceinline__ v16h load_b_f16(const _Float16* Bm, int ldb, int n0, int k0) {
  const int lane = lane_id();
  const int n  = n0 + (lane & 15);
  const int kb = k0 + ((lane >> 4) << 3);
  const _Float16* p = Bm + (size_t)n * ldb + kb;
  v8h b0 = *(const v8h*)(p);
  v8h b1 = *(const v8h*)(p + 16);
  v16h r;
#pragma unroll
  for (int i = 0; i < 8; ++i) { r[i] = b0[i]; r[8 + i] = b1[i]; }
  return r;
}

// ---- Elementwise fp32 -> f16 convert (weights) --------------------------

__global__ void cvt_f32_to_f16(const float* __restrict__ s, _Float16* __restrict__ d, int n) {
  int i = (int)(blockIdx.x * blockDim.x + threadIdx.x) * 4;
  if (i + 3 < n) {
    v4f v = *(const v4f*)(s + i);
#pragma unroll
    for (int j = 0; j < 4; ++j) d[i + j] = (_Float16)v[j];
  }
}

// ---- QKV projection: Y = f16(X @ W^T + b) -------------------------------
// Block = 8 waves (2 M x 4 N), wave tile 64x64 -> block tile 128x256.
// vmode==1 stores V transposed to [b, h, d, s] for the ctx GEMM B operand.
__global__ __launch_bounds__(256) void proj_qkv_kernel(
    const float* __restrict__ X, const _Float16* __restrict__ Wh,
    const float* __restrict__ bias, _Float16* __restrict__ Yrow,
    _Float16* __restrict__ YvT, int vmode) {
  const int wave = (int)(threadIdx.x >> 5);
  const int lane = lane_id();
  const int m0 = (int)blockIdx.y * 128 + (wave >> 2) * 64;
  const int n0 = (int)blockIdx.x * 256 + (wave & 3) * 64;
  v8f acc[4][4] = {};
  for (int k0 = 0; k0 < DM; k0 += 32) {
    v16h a[4], b[4];
#pragma unroll
    for (int i = 0; i < 4; ++i) a[i] = load_a_f32(X, DM, m0 + 16 * i, k0);
#pragma unroll
    for (int j = 0; j < 4; ++j) b[j] = load_b_f16(Wh, DM, n0 + 16 * j, k0);
#pragma unroll
    for (int i = 0; i < 4; ++i)
#pragma unroll
      for (int j = 0; j < 4; ++j)
        acc[i][j] = WMMA_F16(a[i], b[j], acc[i][j]);
  }
#pragma unroll
  for (int i = 0; i < 4; ++i) {
#pragma unroll
    for (int j = 0; j < 4; ++j) {
      const int mb  = m0 + 16 * i + ((lane >> 4) << 3);
      const int col = n0 + 16 * j + (lane & 15);
      const float bv = bias[col];
#pragma unroll
      for (int r = 0; r < 8; ++r) {
        const int row = mb + r;
        const float val = acc[i][j][r] + bv;
        if (vmode == 0) {
          Yrow[(size_t)row * DM + col] = (_Float16)val;
        } else {
          const int bidx = row >> 11, s = row & (SS - 1);
          const int h = col >> 6,     d = col & (DK - 1);
          YvT[((size_t)(bidx * NH + h) * DK + d) * SS + s] = (_Float16)val;
        }
      }
    }
  }
}

// ---- Fused attention per (b,h): scores + softmax + attn-write + ctx -----
// Two-pass flash style: pass A gets per-row max/sum online; pass B recomputes
// the score tiles, writes normalized attn (fp32, non-temporal) and accumulates
// ctx = p @ v via WMMA, relaying p C-frag -> A-frag through per-wave LDS.
// Block = 4 waves stacked in M; wave tile = MF*16 rows.
__global__ __launch_bounds__(128) void fused_attn_kernel(
    const _Float16* __restrict__ qh, const _Float16* __restrict__ kh,
    const _Float16* __restrict__ vT, float* __restrict__ attn,
    _Float16* __restrict__ ctxh) {
  __shared__ _Float16 pbuf[4][16 * 40];   // per-wave 16x32 f16 tile, stride 40
  const int bh = (int)blockIdx.z;
  const int bidx = bh >> 4, h = bh & 15;
  const _Float16* qb = qh + (size_t)bidx * SS * DM + h * DK;
  const _Float16* kb = kh + (size_t)bidx * SS * DM + h * DK;
  const _Float16* vb = vT + (size_t)bh * DK * SS;
  float* ab = attn + (size_t)bh * SS * SS;
  const int wave = (int)(threadIdx.x >> 5);
  const int lane = lane_id();
  const int m0 = (int)blockIdx.x * (MF * 16 * 4) + wave * (MF * 16);
  const int colh = lane & 15;
  const int rsel = (lane >> 4) << 3;

  // q fragments stay resident: MF x (K=64 over two 32-deep frags)
  v16h aq[MF][2];
#pragma unroll
  for (int i = 0; i < MF; ++i)
#pragma unroll
    for (int kk = 0; kk < 2; ++kk)
      aq[i][kk] = load_a_f16(qb, DM, m0 + 16 * i, 32 * kk);

  float rmax[MF][8], rsum[MF][8];
#pragma unroll
  for (int i = 0; i < MF; ++i)
#pragma unroll
    for (int r = 0; r < 8; ++r) { rmax[i][r] = -3.0e38f; rsum[i][r] = 0.f; }

  // ---------------- pass A: online row max / exp-sum ----------------
  for (int kt = 0; kt < SS; kt += 32) {
    v16h bk[2][2];
#pragma unroll
    for (int j = 0; j < 2; ++j)
#pragma unroll
      for (int kk = 0; kk < 2; ++kk)
        bk[j][kk] = load_b_f16(kb, DM, kt + 16 * j, 32 * kk);
#pragma unroll
    for (int i = 0; i < MF; ++i) {
      v8f s0 = {}, s1 = {};
      s0 = WMMA_F16(aq[i][0], bk[0][0], s0);
      s0 = WMMA_F16(aq[i][1], bk[0][1], s0);
      s1 = WMMA_F16(aq[i][0], bk[1][0], s1);
      s1 = WMMA_F16(aq[i][1], bk[1][1], s1);
#pragma unroll
      for (int r = 0; r < 8; ++r) { s0[r] *= 0.125f; s1[r] *= 0.125f; }
      v8f tm;
#pragma unroll
      for (int r = 0; r < 8; ++r) tm[r] = fmaxf(s0[r], s1[r]);
#pragma unroll
      for (int d = 1; d < 16; d <<= 1)
#pragma unroll
        for (int r = 0; r < 8; ++r) tm[r] = fmaxf(tm[r], __shfl_xor(tm[r], d));
      float corr[8];
#pragma unroll
      for (int r = 0; r < 8; ++r) {
        const float nm = fmaxf(rmax[i][r], tm[r]);
        corr[r] = __expf(rmax[i][r] - nm);
        rmax[i][r] = nm;
      }
      v8f ts;
#pragma unroll
      for (int r = 0; r < 8; ++r)
        ts[r] = __expf(s0[r] - rmax[i][r]) + __expf(s1[r] - rmax[i][r]);
#pragma unroll
      for (int d = 1; d < 16; d <<= 1)
#pragma unroll
        for (int r = 0; r < 8; ++r) ts[r] += __shfl_xor(ts[r], d);
#pragma unroll
      for (int r = 0; r < 8; ++r) rsum[i][r] = rsum[i][r] * corr[r] + ts[r];
    }
  }

  float rinv[MF][8];
#pragma unroll
  for (int i = 0; i < MF; ++i)
#pragma unroll
    for (int r = 0; r < 8; ++r) rinv[i][r] = 1.0f / rsum[i][r];

  // ---------------- pass B: write attn + accumulate ctx ----------------
  v8f acc[MF][4] = {};
  _Float16* tb = &pbuf[wave][0];
  for (int kt = 0; kt < SS; kt += 32) {
    v16h bk[2][2];
#pragma unroll
    for (int j = 0; j < 2; ++j)
#pragma unroll
      for (int kk = 0; kk < 2; ++kk)
        bk[j][kk] = load_b_f16(kb, DM, kt + 16 * j, 32 * kk);
    v16h bv[4];
#pragma unroll
    for (int jd = 0; jd < 4; ++jd) bv[jd] = load_b_f16(vb, SS, 16 * jd, kt);
#pragma unroll
    for (int i = 0; i < MF; ++i) {
      v8f s0 = {}, s1 = {};
      s0 = WMMA_F16(aq[i][0], bk[0][0], s0);
      s0 = WMMA_F16(aq[i][1], bk[0][1], s0);
      s1 = WMMA_F16(aq[i][0], bk[1][0], s1);
      s1 = WMMA_F16(aq[i][1], bk[1][1], s1);
      v8f p0, p1;
#pragma unroll
      for (int r = 0; r < 8; ++r) {
        p0[r] = __expf(s0[r] * 0.125f - rmax[i][r]) * rinv[i][r];
        p1[r] = __expf(s1[r] * 0.125f - rmax[i][r]) * rinv[i][r];
      }
      // mandated attn output: fp32, streaming (never re-read) -> non-temporal
      const int rowb = m0 + 16 * i + rsel;
#pragma unroll
      for (int r = 0; r < 8; ++r)
        __builtin_nontemporal_store(p0[r], ab + (size_t)(rowb + r) * SS + kt + colh);
#pragma unroll
      for (int r = 0; r < 8; ++r)
        __builtin_nontemporal_store(p1[r], ab + (size_t)(rowb + r) * SS + kt + 16 + colh);
      // C-frag -> A-frag relayout through per-wave LDS tile
#pragma unroll
      for (int r = 0; r < 8; ++r) tb[(rsel + r) * 40 + colh]      = (_Float16)p0[r];
#pragma unroll
      for (int r = 0; r < 8; ++r) tb[(rsel + r) * 40 + 16 + colh] = (_Float16)p1[r];
      asm volatile("s_wait_dscnt 0x0" ::: "memory");   // same-wave LDS RAW
      v16h pa = load_a_lds(tb);
#pragma unroll
      for (int jd = 0; jd < 4; ++jd) acc[i][jd] = WMMA_F16(pa, bv[jd], acc[i][jd]);
      asm volatile("s_wait_dscnt 0x0" ::: "memory");   // drain loads before reuse
    }
  }

  // ctx epilogue: f16 into [b, s, h*64+d] (concat layout for the out GEMM)
#pragma unroll
  for (int i = 0; i < MF; ++i) {
#pragma unroll
    for (int jd = 0; jd < 4; ++jd) {
      const int mb = m0 + 16 * i + rsel;
      const int d  = 16 * jd + colh;
#pragma unroll
      for (int r = 0; r < 8; ++r)
        ctxh[((size_t)bidx * SS + (mb + r)) * DM + h * DK + d] = (_Float16)acc[i][jd][r];
    }
  }
}

// ---- Output projection: out = ctx @ Wo^T + bo (fp32 out) ----------------
__global__ __launch_bounds__(256) void out_proj_kernel(
    const _Float16* __restrict__ ctxh, const _Float16* __restrict__ Woh,
    const float* __restrict__ bo, float* __restrict__ outp) {
  const int wave = (int)(threadIdx.x >> 5);
  const int lane = lane_id();
  const int m0 = (int)blockIdx.y * 128 + (wave >> 2) * 64;
  const int n0 = (int)blockIdx.x * 256 + (wave & 3) * 64;
  v8f acc[4][4] = {};
  for (int k0 = 0; k0 < DM; k0 += 32) {
    v16h a[4], b[4];
#pragma unroll
    for (int i = 0; i < 4; ++i) a[i] = load_a_f16(ctxh, DM, m0 + 16 * i, k0);
#pragma unroll
    for (int j = 0; j < 4; ++j) b[j] = load_b_f16(Woh, DM, n0 + 16 * j, k0);
#pragma unroll
    for (int i = 0; i < 4; ++i)
#pragma unroll
      for (int j = 0; j < 4; ++j)
        acc[i][j] = WMMA_F16(a[i], b[j], acc[i][j]);
  }
#pragma unroll
  for (int i = 0; i < 4; ++i) {
#pragma unroll
    for (int j = 0; j < 4; ++j) {
      const int mb  = m0 + 16 * i + ((lane >> 4) << 3);
      const int col = n0 + 16 * j + (lane & 15);
      const float bv = bo[col];
#pragma unroll
      for (int r = 0; r < 8; ++r)
        outp[(size_t)(mb + r) * DM + col] = acc[i][j][r] + bv;
    }
  }
}

// -------------------------------------------------------------------------

extern "C" void kernel_launch(void* const* d_in, const int* in_sizes, int n_in,
                              void* d_out, int out_size, void* d_ws, size_t ws_size,
                              hipStream_t stream) {
  (void)in_sizes; (void)n_in; (void)out_size; (void)ws_size;
  const float* Q  = (const float*)d_in[0];
  const float* K  = (const float*)d_in[1];
  const float* V  = (const float*)d_in[2];
  const float* Wq = (const float*)d_in[3];
  const float* bq = (const float*)d_in[4];
  const float* Wk = (const float*)d_in[5];
  const float* bk = (const float*)d_in[6];
  const float* Wv = (const float*)d_in[7];
  const float* bv = (const float*)d_in[8];
  const float* Wo = (const float*)d_in[9];
  const float* bo = (const float*)d_in[10];

  float* outp = (float*)d_out;                        // [8192, 1024]
  float* attn = outp + (size_t)MROWS * DM;            // [64, 2048, 2048]

  char* ws = (char*)d_ws;
  _Float16* Wq_h  = (_Float16*)(ws + ((size_t)0  << 20));
  _Float16* Wk_h  = (_Float16*)(ws + ((size_t)2  << 20));
  _Float16* Wv_h  = (_Float16*)(ws + ((size_t)4  << 20));
  _Float16* Wo_h  = (_Float16*)(ws + ((size_t)6  << 20));
  _Float16* q_h   = (_Float16*)(ws + ((size_t)8  << 20));  // 16 MB
  _Float16* k_h   = (_Float16*)(ws + ((size_t)24 << 20));  // 16 MB
  _Float16* vT_h  = (_Float16*)(ws + ((size_t)40 << 20));  // 16 MB  [b,h,d,s]
  _Float16* ctx_h = (_Float16*)(ws + ((size_t)56 << 20));  // 16 MB

  const int WN = DM * DM;
  cvt_f32_to_f16<<<WN / 1024, 256, 0, stream>>>(Wq, Wq_h, WN);
  cvt_f32_to_f16<<<WN / 1024, 256, 0, stream>>>(Wk, Wk_h, WN);
  cvt_f32_to_f16<<<WN / 1024, 256, 0, stream>>>(Wv, Wv_h, WN);
  cvt_f32_to_f16<<<WN / 1024, 256, 0, stream>>>(Wo, Wo_h, WN);

  dim3 pg(DM / 256, MROWS / 128, 1);
  proj_qkv_kernel<<<pg, 256, 0, stream>>>(Q, Wq_h, bq, q_h, nullptr, 0);
  proj_qkv_kernel<<<pg, 256, 0, stream>>>(K, Wk_h, bk, k_h, nullptr, 0);
  proj_qkv_kernel<<<pg, 256, 0, stream>>>(V, Wv_h, bv, nullptr, vT_h, 1);

  dim3 fg(SS / (MF * 16 * 4), 1, BB * NH);   // 16 row-blocks x 64 (b,h)
  fused_attn_kernel<<<fg, 128, 0, stream>>>(q_h, k_h, vT_h, attn, ctx_h);

  out_proj_kernel<<<pg, 256, 0, stream>>>(ctx_h, Wo_h, bo, outp);
}
